// Commnet_88699664597678
// MI455X (gfx1250) — matrix operand order repaired and checked by reading, not compiled, not run on hardware
//
#include <hip/hip_runtime.h>
#include <hip/hip_bf16.h>

#define N_WORDS   32000
#define EDIM      256
#define N_AGENTS  131072
#define BAG       8
#define N_BATCHES 4096
#define N_ACTIONS 16

typedef __attribute__((ext_vector_type(16))) _Float16 v16h;
typedef __attribute__((ext_vector_type(8)))  float    v8f;
typedef __attribute__((ext_vector_type(4)))  _Float16 h4;

// ---------------------------------------------------------------------------
// 1) EmbeddingBag mean: emb[a] = (1/8) * sum_j table[x[a*8+j]]
//    4 agents per 256-thread block, 64 threads (float4 each) per agent.
// ---------------------------------------------------------------------------
__global__ __launch_bounds__(256) void embed_kernel(const int* __restrict__ x,
                                                    const float* __restrict__ table,
                                                    float* __restrict__ emb) {
    __shared__ int w[4 * BAG];
    const int tid = threadIdx.x;
    const int agent0 = blockIdx.x * 4;
    if (tid < 4 * BAG) w[tid] = x[agent0 * BAG + tid];
    __syncthreads();
    const int la = tid >> 6;             // local agent 0..3
    const int d4 = (tid & 63) << 2;      // dim, step 4
    const int* wa = &w[la * BAG];
    float4 acc = {0.f, 0.f, 0.f, 0.f};
#pragma unroll
    for (int j = 0; j < BAG; ++j) {
        const float4 v = *reinterpret_cast<const float4*>(&table[(long)wa[j] * EDIM + d4]);
        acc.x += v.x; acc.y += v.y; acc.z += v.z; acc.w += v.w;
    }
    const float s = 1.0f / (float)BAG;
    float4 o = {acc.x * s, acc.y * s, acc.z * s, acc.w * s};
    *reinterpret_cast<float4*>(&emb[(long)(agent0 + la) * EDIM + d4]) = o;
}

// ---------------------------------------------------------------------------
// 2) Segment bounds on the sorted batch_idx: bstart[b] = lower_bound(bidx, b)
// ---------------------------------------------------------------------------
__global__ void bounds_kernel(const int* __restrict__ bidx, int* __restrict__ bstart) {
    const int b = blockIdx.x * blockDim.x + threadIdx.x;
    if (b > N_BATCHES) return;
    int lo = 0, hi = N_AGENTS;
    while (lo < hi) {
        int mid = (lo + hi) >> 1;
        if (bidx[mid] < b) lo = mid + 1; else hi = mid;
    }
    bstart[b] = lo;
}

// ---------------------------------------------------------------------------
// 3) Segment sum: m[b] = sum of emb rows in [bstart[b], bstart[b+1])
//    grid (N_BATCHES, 2), 128 threads own one dim each; no atomics needed.
// ---------------------------------------------------------------------------
__global__ void segsum_kernel(const float* __restrict__ emb, const int* __restrict__ bstart,
                              float* __restrict__ m) {
    const int b = blockIdx.x;
    const int d = blockIdx.y * 128 + threadIdx.x;
    const int s = bstart[b], e = bstart[b + 1];
    float acc = 0.f;
    for (int a = s; a < e; ++a) acc += emb[(long)a * EDIM + d];
    m[(long)b * EDIM + d] = acc;
}

// ---------------------------------------------------------------------------
// 4) Build f16 A = [emb | (m[b]-emb) / (len[b]-0.99999)]  -> [N_AGENTS, 512]
// ---------------------------------------------------------------------------
__global__ __launch_bounds__(256) void build_a_kernel(const float* __restrict__ emb,
                                                      const float* __restrict__ m,
                                                      const int* __restrict__ bidx,
                                                      const float* __restrict__ blen,
                                                      _Float16* __restrict__ A) {
    const int tid = threadIdx.x;
    const int a = blockIdx.x * 4 + (tid >> 6);
    const int d = (tid & 63) << 2;
    const int b = bidx[a];
    const float s = 1.f / (blen[b] - 0.99999f);
    const float4 e  = *reinterpret_cast<const float4*>(&emb[(long)a * EDIM + d]);
    const float4 mm = *reinterpret_cast<const float4*>(&m[(long)b * EDIM + d]);
    h4 lo = {(_Float16)e.x, (_Float16)e.y, (_Float16)e.z, (_Float16)e.w};
    h4 hi = {(_Float16)((mm.x - e.x) * s), (_Float16)((mm.y - e.y) * s),
             (_Float16)((mm.z - e.z) * s), (_Float16)((mm.w - e.w) * s)};
    *reinterpret_cast<h4*>(&A[(long)a * (2 * EDIM) + d]) = lo;
    *reinterpret_cast<h4*>(&A[(long)a * (2 * EDIM) + EDIM + d]) = hi;
}

// ---------------------------------------------------------------------------
// 5) Convert layer weight W [256, 512] f32 -> f16 (row-major, out x in)
// ---------------------------------------------------------------------------
__global__ void convw_kernel(const float* __restrict__ W, _Float16* __restrict__ Wf) {
    const int i = (blockIdx.x * blockDim.x + threadIdx.x) * 4;
    if (i >= EDIM * 2 * EDIM) return;
    const float4 v = *reinterpret_cast<const float4*>(&W[i]);
    h4 h = {(_Float16)v.x, (_Float16)v.y, (_Float16)v.z, (_Float16)v.w};
    *reinterpret_cast<h4*>(&Wf[i]) = h;
}

// ---------------------------------------------------------------------------
// 6) WMMA GEMM: out[a,o] = relu( sum_k A[a,k] * W[o,k] + bias[o] )
//    A: [M=131072, K=512] f16 row-major; W: [N=256, K=512] f16 row-major.
//    Block tile 128x64, 8 waves -> each wave 32x32 (2x2 v_wmma_f32_16x16x32_f16).
// ---------------------------------------------------------------------------
#define BM 128
#define BN 64
#define BK 32
#define KTOT 512
#define LPITCH 40   // halves per LDS row: 32 data + 8 pad (80 B, 16B-aligned)

__global__ __launch_bounds__(256) void gemm_kernel(const _Float16* __restrict__ A,
                                                   const _Float16* __restrict__ W,
                                                   const float* __restrict__ bias,
                                                   float* __restrict__ out) {
    __shared__ _Float16 sA[BM * LPITCH];
    __shared__ _Float16 sB[BN * LPITCH];
    const int tid  = threadIdx.x;
    const int lane = tid & 31;
    const int wave = tid >> 5;
    const int wm = wave & 3;               // row block (32 rows)
    const int wn = wave >> 2;              // col block (32 cols)
    const long tileM = (long)blockIdx.x * BM;
    const int  tileN = blockIdx.y * BN;

    const int lrow  = lane & 15;
    const int hiH   = lane >> 4;           // 0 for lanes 0-15, 1 for 16-31
    const int koffA = hiH ? 8 : 0;         // A: second half-wave takes K+8 halves
    const int koffB = hiH ? 8 : 0;         // B: second half-wave takes K+8 halves

    union Frag { uint4 u[2]; v16h v; };
    v8f acc[2][2] = {};

    for (int kk = 0; kk < KTOT; kk += BK) {
        // Stage A tile: 128 rows x 32 halves (2 x 16B per thread over 2 iters)
#pragma unroll
        for (int it = 0; it < 2; ++it) {
            const int idx = it * 256 + tid;
            const int row = idx >> 2;
            const int seg = idx & 3;
            const uint4 v = *reinterpret_cast<const uint4*>(
                &A[(tileM + row) * (long)KTOT + kk + seg * 8]);
            *reinterpret_cast<uint4*>(&sA[row * LPITCH + seg * 8]) = v;
        }
        // Stage B tile: 64 rows x 32 halves (1 x 16B per thread)
        {
            const int row = tid >> 2;
            const int seg = tid & 3;
            const uint4 v = *reinterpret_cast<const uint4*>(
                &W[(tileN + row) * (long)KTOT + kk + seg * 8]);
            *reinterpret_cast<uint4*>(&sB[row * LPITCH + seg * 8]) = v;
        }
        // Prefetch next K-slab of the A tile into cache (global_prefetch_b8)
        if (kk + BK < KTOT) {
            __builtin_prefetch(&A[(tileM + (tid >> 1)) * (long)KTOT + kk + BK + (tid & 1) * 16], 0, 3);
        }
        __syncthreads();

        Frag fa[2], fb[2];
#pragma unroll
        for (int i = 0; i < 2; ++i) {
            const int aBase = (wm * 32 + i * 16 + lrow) * LPITCH + koffA;
            fa[i].u[0] = *reinterpret_cast<const uint4*>(&sA[aBase]);        // K koff+0..7
            fa[i].u[1] = *reinterpret_cast<const uint4*>(&sA[aBase + 16]);   // K koff+16..23
        }
#pragma unroll
        for (int j = 0; j < 2; ++j) {
            const int bBase = (wn * 32 + j * 16 + lrow) * LPITCH + koffB;
            fb[j].u[0] = *reinterpret_cast<const uint4*>(&sB[bBase]);        // K koff..+7
            fb[j].u[1] = *reinterpret_cast<const uint4*>(&sB[bBase + 16]);   // K koff+16..+23
        }
#pragma unroll
        for (int i = 0; i < 2; ++i)
#pragma unroll
            for (int j = 0; j < 2; ++j)
                acc[i][j] = __builtin_amdgcn_wmma_f32_16x16x32_f16(
                    false, fa[i].v, false, fb[j].v, (short)0, acc[i][j], false, false);
        __syncthreads();
    }

    // Epilogue: C/D layout — VGPR r holds M = r + 8*(lane>=16), N = lane&15
    const int rOff = hiH ? 8 : 0;
#pragma unroll
    for (int j = 0; j < 2; ++j) {
        const int col = tileN + wn * 32 + j * 16 + lrow;
        const float bv = bias[col];
#pragma unroll
        for (int i = 0; i < 2; ++i) {
            const long rowBase = tileM + wm * 32 + i * 16 + rOff;
#pragma unroll
            for (int r = 0; r < 8; ++r) {
                float v = acc[i][j][r] + bv;
                out[(rowBase + r) * EDIM + col] = v > 0.f ? v : 0.f;
            }
        }
    }
}

// ---------------------------------------------------------------------------
// 7) Final layer fused: m[b] = segsum(emb); logits = m @ Wh^T + bh; log_softmax
// ---------------------------------------------------------------------------
__global__ __launch_bounds__(256) void head_kernel(const float* __restrict__ emb,
                                                   const int* __restrict__ bstart,
                                                   const float* __restrict__ Wh,
                                                   const float* __restrict__ bh,
                                                   float* __restrict__ outp) {
    __shared__ float sm[EDIM];
    __shared__ float sl[N_ACTIONS];
    const int b = blockIdx.x;
    const int tid = threadIdx.x;
    const int s = bstart[b], e = bstart[b + 1];
    float acc = 0.f;
    for (int a = s; a < e; ++a) acc += emb[(long)a * EDIM + tid];
    sm[tid] = acc;
    __syncthreads();
    if (tid < N_ACTIONS) {
        float dot = bh[tid];
        const float* w = &Wh[tid * EDIM];
#pragma unroll 8
        for (int d = 0; d < EDIM; ++d) dot += w[d] * sm[d];
        sl[tid] = dot;
    }
    __syncthreads();
    if (tid < N_ACTIONS) {
        float mx = sl[0];
#pragma unroll
        for (int o = 1; o < N_ACTIONS; ++o) mx = fmaxf(mx, sl[o]);
        float sum = 0.f;
#pragma unroll
        for (int o = 0; o < N_ACTIONS; ++o) sum += expf(sl[o] - mx);
        outp[(long)b * N_ACTIONS + tid] = sl[tid] - mx - logf(sum);
    }
}

// ---------------------------------------------------------------------------
extern "C" void kernel_launch(void* const* d_in, const int* in_sizes, int n_in,
                              void* d_out, int out_size, void* d_ws, size_t ws_size,
                              hipStream_t stream) {
    const int*   x     = (const int*)  d_in[0];
    // d_in[1] item_starts: uniform bags of 8, not needed
    const int*   bidx  = (const int*)  d_in[2];
    const float* blen  = (const float*)d_in[3];
    const float* table = (const float*)d_in[4];
    const float* W0    = (const float*)d_in[5];
    const float* b0    = (const float*)d_in[6];
    const float* W1    = (const float*)d_in[7];
    const float* b1    = (const float*)d_in[8];
    const float* Whd   = (const float*)d_in[9];
    const float* bh    = (const float*)d_in[10];
    float* outp = (float*)d_out;

    char* ws = (char*)d_ws;
    size_t off = 0;
    auto alloc = [&](size_t bytes) -> void* {
        void* p = ws + off;
        off = (off + bytes + 255) & ~(size_t)255;
        return p;
    };
    float*    emb    = (float*)   alloc((size_t)N_AGENTS * EDIM * sizeof(float));
    _Float16* aF16   = (_Float16*)alloc((size_t)N_AGENTS * 2 * EDIM * sizeof(_Float16));
    float*    m      = (float*)   alloc((size_t)N_BATCHES * EDIM * sizeof(float));
    _Float16* wF16   = (_Float16*)alloc((size_t)EDIM * 2 * EDIM * sizeof(_Float16));
    int*      bstart = (int*)     alloc((size_t)(N_BATCHES + 1) * sizeof(int));
    (void)ws_size; (void)in_sizes; (void)n_in; (void)out_size;

    embed_kernel<<<N_AGENTS / 4, 256, 0, stream>>>(x, table, emb);
    bounds_kernel<<<(N_BATCHES + 1 + 255) / 256, 256, 0, stream>>>(bidx, bstart);

    for (int layer = 0; layer < 2; ++layer) {
        const float* W  = layer ? W1 : W0;
        const float* bb = layer ? b1 : b0;
        segsum_kernel<<<dim3(N_BATCHES, 2), 128, 0, stream>>>(emb, bstart, m);
        convw_kernel<<<(EDIM * 2 * EDIM / 4 + 255) / 256, 256, 0, stream>>>(W, wF16);
        build_a_kernel<<<N_AGENTS / 4, 256, 0, stream>>>(emb, m, bidx, blen, aF16);
        gemm_kernel<<<dim3(N_AGENTS / BM, EDIM / BN), 256, 0, stream>>>(aF16, wF16, bb, emb);
    }

    head_kernel<<<N_BATCHES, 256, 0, stream>>>(emb, bstart, Whd, bh, outp);
}
